// VectorQuantizer_36764920054252
// MI455X (gfx1250) — matrix operand order repaired and checked by reading, not compiled, not run on hardware
//
#include <hip/hip_runtime.h>
#include <stdint.h>

// ---------------------------------------------------------------------------
// VQ nearest-centroid quantizer for MI455X (gfx1250).
// N=32768 rows, D=512, K=8192 centroids.
// Core: bf16 WMMA streaming argmin-GEMM. Each wave owns 32 rows so every
// LDS B fragment feeds two WMMAs; codebook tiles staged with double-buffered
// async global->LDS copies; B fragments software-pipelined across subtiles.
// ---------------------------------------------------------------------------

typedef __bf16 bf16;
typedef __attribute__((ext_vector_type(4)))  bf16  v4bf;
typedef __attribute__((ext_vector_type(8)))  bf16  v8bf;
typedef __attribute__((ext_vector_type(16))) bf16  v16bf;
typedef __attribute__((ext_vector_type(8)))  float v8f;
typedef __attribute__((ext_vector_type(4)))  int   v4i;

#define VQ_D 512
#define VQ_K 8192
#define VQ_N 32768            // 16 * 2048
#define COMMIT_SCALE 1.25f    // (1 + COMMITMENT) * (q - x)^2

#define XT_ROWS   256         // rows per block
#define XT_STRIDE 520         // x-tile LDS row stride in bf16 (512 + 8 pad)
#define XT_BYTES  (XT_ROWS * XT_STRIDE * 2)            // 266240
#define BT_K      64          // k-depth per B stage
#define BT_STRIDE 72          // B-tile LDS row stride in bf16 (64 + 8 pad)
#define BSTAGE_ELEMS (128 * BT_STRIDE)                 // 9216
#define SMEM_BYTES (XT_BYTES + 2 * BSTAGE_ELEMS * 2)   // 266240+36864 = 303104
#define N_STAGES  ((VQ_K / 128) * (VQ_D / BT_K))       // 64 col tiles * 8 = 512

// ---------------------------------------------------------------------------
// gfx1250 async global->LDS copy (ASYNCcnt-tracked), with sync fallback.
// Builtin prototype: (v4i AS1*, v4i AS3*, Ii offset, Ii cpol).
// ---------------------------------------------------------------------------
#if defined(__HIP_DEVICE_COMPILE__) && __has_builtin(__builtin_amdgcn_global_load_async_to_lds_b128)
#define HAVE_ASYNC_LDS 1
#endif

typedef __attribute__((address_space(1))) v4i gv4i_t;  // global int4
typedef __attribute__((address_space(3))) v4i lv4i_t;  // LDS int4

__device__ __forceinline__ void async_copy_b128(const bf16* gsrc, bf16* ldst) {
#ifdef HAVE_ASYNC_LDS
  __builtin_amdgcn_global_load_async_to_lds_b128(
      (gv4i_t*)(uintptr_t)gsrc,
      (lv4i_t*)(uint32_t)(uintptr_t)ldst,  // low 32 bits of flat shared ptr = LDS addr
      0, 0);
#else
  *(int4*)ldst = *(const int4*)gsrc;
#endif
}

__device__ __forceinline__ void wait_async_le4() {
#ifdef HAVE_ASYNC_LDS
#if __has_builtin(__builtin_amdgcn_s_wait_asynccnt)
  __builtin_amdgcn_s_wait_asynccnt(4);
#else
  asm volatile("s_wait_asynccnt 0x4" ::: "memory");
#endif
#endif
}

__device__ __forceinline__ void wait_async_le0() {
#ifdef HAVE_ASYNC_LDS
#if __has_builtin(__builtin_amdgcn_s_wait_asynccnt)
  __builtin_amdgcn_s_wait_asynccnt(0);
#else
  asm volatile("s_wait_asynccnt 0x0" ::: "memory");
#endif
#endif
}

// ---------------------------------------------------------------------------
// f32 -> bf16 conversion of inputs (vectorized 4-wide)
// ---------------------------------------------------------------------------
__global__ __launch_bounds__(256) void vq_cvt_x_kernel(const float* __restrict__ x,
                                                       bf16* __restrict__ xb) {
  long i = (long)blockIdx.x * 256 + threadIdx.x;   // over N*D/4
  float4 v = ((const float4*)x)[i];
  v4bf o = { (bf16)v.x, (bf16)v.y, (bf16)v.z, (bf16)v.w };
  ((v4bf*)xb)[i] = o;
}

// f32 -> bf16 codebook conversion + pass-through copy to output slot 3
__global__ __launch_bounds__(256) void vq_cvt_c_kernel(const float* __restrict__ c,
                                                       bf16* __restrict__ cb,
                                                       float* __restrict__ outCB) {
  long i = (long)blockIdx.x * 256 + threadIdx.x;   // over K*D/4
  float4 v = ((const float4*)c)[i];
  v4bf o = { (bf16)v.x, (bf16)v.y, (bf16)v.z, (bf16)v.w };
  ((v4bf*)cb)[i] = o;
  ((float4*)outCB)[i] = v;
}

// ---------------------------------------------------------------------------
// ||c_k||^2 per centroid (one wave32 per centroid) + counts = 1.0 output
// ---------------------------------------------------------------------------
__global__ __launch_bounds__(256) void vq_cnorm_kernel(const float* __restrict__ c,
                                                       float* __restrict__ cnorm,
                                                       float* __restrict__ outCounts) {
  int wv   = (int)((blockIdx.x * 256 + threadIdx.x) >> 5);  // centroid id
  int lane = threadIdx.x & 31;
  const float* row = c + (size_t)wv * VQ_D;
  float s = 0.0f;
  #pragma unroll
  for (int j = 0; j < VQ_D / 32; ++j) {
    float v = row[j * 32 + lane];          // coalesced across the wave
    s += v * v;
  }
  #pragma unroll
  for (int off = 16; off >= 1; off >>= 1) s += __shfl_xor(s, off, 32);
  if (lane == 0) {
    cnorm[wv] = s;
    outCounts[wv] = 1.0f;                  // train=False: EMA counters stay ones
  }
}

// ---------------------------------------------------------------------------
// Streaming argmin GEMM:  dist = ||c||^2 - 2 * (x . c)   (||x||^2 dropped)
// 128 blocks x 256 threads (8 waves). Block owns 256 rows; wave owns 32 rows
// (two 16-row WMMA blocks -> each B fragment feeds two WMMAs).
// ---------------------------------------------------------------------------
__global__ __launch_bounds__(256) void vq_argmin_kernel(const bf16* __restrict__ xb,
                                                        const bf16* __restrict__ cb,
                                                        const float* __restrict__ cnorm,
                                                        int* __restrict__ bestIdx) {
  extern __shared__ char smem[];
  bf16* xl     = (bf16*)smem;                     // [256][XT_STRIDE]
  bf16* bstage = (bf16*)(smem + XT_BYTES);        // 2 x [128][BT_STRIDE]

  const int t    = threadIdx.x;
  const int wave = t >> 5;
  const int lane = t & 31;
  const int h    = lane >> 4;                     // lane half (K/M split)
  const int ln   = lane & 15;
  const long rowBase = (long)blockIdx.x * XT_ROWS;

  // ---- stage the 256x512 bf16 x-tile into LDS (16B chunks, coalesced) ----
  #pragma unroll 4
  for (int i = 0; i < 64; ++i) {
    int chunk = t + i * 256;                      // 0..16383
    int r  = chunk >> 6;                          // row 0..255
    int ko = (chunk & 63) << 3;                   // k offset, multiple of 8
    int4 v = *(const int4*)(xb + (rowBase + r) * VQ_D + ko);
    *(int4*)(xl + r * XT_STRIDE + ko) = v;
  }

  // Per-thread fixed mapping for B-tile staging: 4 chunks per stage.
  // chunk = i*256 + t ->  col = (t>>3) + i*32,  ko = (t&7)*8  (constant).
  const int colb0 = t >> 3;
  const int kob   = (t & 7) << 3;

  // Issue stage 0 while the x-tile settles.
  {
    const bf16* gbase = cb + (size_t)colb0 * VQ_D + kob;    // ct=0, kbase=0
    bf16* lbase = bstage + colb0 * BT_STRIDE + kob;
    #pragma unroll
    for (int i = 0; i < 4; ++i)
      async_copy_b128(gbase + (size_t)i * 32 * VQ_D, lbase + i * 32 * BT_STRIDE);
  }

  const int m0 = wave * 32;                       // wave's first row inside tile
  float minv[16];
  int   mini[16];
  #pragma unroll
  for (int v = 0; v < 16; ++v) { minv[v] = 3.4e38f; mini[v] = 0; }

  v8f acc[16];                                    // [n0][rb] -> acc[n0*2+rb]

  for (int s = 0; s < N_STAGES; ++s) {
    // ---- issue async staging for stage s+1 into the other buffer ----
    if (s + 1 < N_STAGES) {
      const int s1  = s + 1;
      const int ct1 = s1 >> 3;
      const int kb1 = (s1 & 7) * BT_K;
      const bf16* gbase = cb + (size_t)(ct1 * 128 + colb0) * VQ_D + kb1 + kob;
      bf16* lbase = bstage + (s1 & 1) * BSTAGE_ELEMS + colb0 * BT_STRIDE + kob;
      #pragma unroll
      for (int i = 0; i < 4; ++i)
        async_copy_b128(gbase + (size_t)i * 32 * VQ_D, lbase + i * 32 * BT_STRIDE);
      wait_async_le4();                           // in-order: stage s has landed
    } else {
      wait_async_le0();
    }
    __syncthreads();                              // all waves' stage-s data visible

    const int ct    = s >> 3;
    const int kbase = (s & 7) * BT_K;
    const bf16* bbuf = bstage + (s & 1) * BSTAGE_ELEMS;

    if ((s & 7) == 0) {
      v8f zero = {};
      #pragma unroll
      for (int q = 0; q < 16; ++q) acc[q] = zero;
    }

    // ---- A fragments: 2 row blocks x 2 k-steps, kept in registers ----
    v16bf af[2][2];
    #pragma unroll
    for (int rb = 0; rb < 2; ++rb) {
      const bf16* arow = xl + (m0 + rb * 16 + ln) * XT_STRIDE + kbase;
      #pragma unroll
      for (int kki = 0; kki < 2; ++kki) {
        const bf16* ab = arow + kki * 32 + 8 * h;
        v8bf alo = *(const v8bf*)ab;              // k in [8h, 8h+8)
        v8bf ahi = *(const v8bf*)(ab + 16);       // k in [16+8h, 16+8h+8)
        af[rb][kki] = __builtin_shufflevector(alo, ahi, 0, 1, 2, 3, 4, 5, 6, 7,
                                                        8, 9, 10, 11, 12, 13, 14, 15);
      }
    }

    // ---- B fragments software-pipelined one subtile ahead: the 4 loads of
    //      subtile n0+1 are in flight while the 4 WMMAs of n0 execute. ----
    const bf16* bwave = bbuf + ln * BT_STRIDE + 8 * h;
    v16bf bcur[2];
    {
      #pragma unroll
      for (int kki = 0; kki < 2; ++kki) {
        v8bf blo = *(const v8bf*)(bwave + kki * 32);
        v8bf bhi = *(const v8bf*)(bwave + kki * 32 + 16);
        bcur[kki] = __builtin_shufflevector(blo, bhi, 0, 1, 2, 3, 4, 5, 6, 7,
                                                      8, 9, 10, 11, 12, 13, 14, 15);
      }
    }
    #pragma unroll
    for (int n0 = 0; n0 < 8; ++n0) {
      v16bf bnxt[2];
      if (n0 < 7) {
        const bf16* bb = bwave + (n0 + 1) * 16 * BT_STRIDE;
        #pragma unroll
        for (int kki = 0; kki < 2; ++kki) {
          v8bf blo = *(const v8bf*)(bb + kki * 32);
          v8bf bhi = *(const v8bf*)(bb + kki * 32 + 16);
          bnxt[kki] = __builtin_shufflevector(blo, bhi, 0, 1, 2, 3, 4, 5, 6, 7,
                                                        8, 9, 10, 11, 12, 13, 14, 15);
        }
      }
      #pragma unroll
      for (int kki = 0; kki < 2; ++kki)
        #pragma unroll
        for (int rb = 0; rb < 2; ++rb)
          acc[n0 * 2 + rb] = __builtin_amdgcn_wmma_f32_16x16x32_bf16(
              /*neg_a=*/false, af[rb][kki], /*neg_b=*/false, bcur[kki],
              /*c_mod=*/(short)0, acc[n0 * 2 + rb],
              /*reuse_a=*/false, /*reuse_b=*/false);
      if (n0 < 7) {
        bcur[0] = bnxt[0];
        bcur[1] = bnxt[1];
      }
    }

    // Column tile finished: fold into the running per-lane (min, argmin).
    // C layout: lane l holds N = l%16; VGPR v holds M = rb*16 + v + 8*(l/16).
    if ((s & 7) == 7) {
      #pragma unroll
      for (int n0 = 0; n0 < 8; ++n0) {
        const int col = ct * 128 + n0 * 16 + ln;
        const float cn = cnorm[col];
        #pragma unroll
        for (int rb = 0; rb < 2; ++rb)
          #pragma unroll
          for (int v = 0; v < 8; ++v) {
            float d = cn - 2.0f * acc[n0 * 2 + rb][v];
            const int q = rb * 8 + v;
            if (d < minv[q] || (d == minv[q] && col < mini[q])) {
              minv[q] = d;
              mini[q] = col;
            }
          }
      }
    }
    __syncthreads();                              // reads done before overwrite
  }

  // Cross-lane min over the 16 lanes sharing the same M rows (xor stays in half).
  #pragma unroll
  for (int off = 8; off >= 1; off >>= 1) {
    #pragma unroll
    for (int q = 0; q < 16; ++q) {
      float ov = __shfl_xor(minv[q], off, 32);
      int   oi = __shfl_xor(mini[q], off, 32);
      if (ov < minv[q] || (ov == minv[q] && oi < mini[q])) {
        minv[q] = ov;
        mini[q] = oi;
      }
    }
  }
  if (ln == 0) {
    #pragma unroll
    for (int rb = 0; rb < 2; ++rb)
      #pragma unroll
      for (int v = 0; v < 8; ++v)
        bestIdx[rowBase + m0 + rb * 16 + 8 * h + v] = mini[rb * 8 + v];
  }
}

// ---------------------------------------------------------------------------
// Gather + outputs: quantized = C[idx], loss = 1.25*(q-x)^2, idx as float.
// ---------------------------------------------------------------------------
__global__ __launch_bounds__(256) void vq_finalize_kernel(const float* __restrict__ x,
                                                          const float* __restrict__ c,
                                                          const int* __restrict__ bestIdx,
                                                          float* __restrict__ outQ,
                                                          float* __restrict__ outL,
                                                          float* __restrict__ outIdx) {
  long i = (long)blockIdx.x * 256 + threadIdx.x;   // over N*D/4
  long n = i >> 7;                                  // i / (D/4)
  int  off4 = (int)(i & 127);
  int  idx = bestIdx[n];
  float4 q = ((const float4*)(c + (size_t)idx * VQ_D))[off4];
  float4 xv = ((const float4*)x)[i];
  ((float4*)outQ)[i] = q;                           // straight-through == quantized
  float4 l;
  float dx = q.x - xv.x, dy = q.y - xv.y, dz = q.z - xv.z, dw = q.w - xv.w;
  l.x = COMMIT_SCALE * dx * dx;
  l.y = COMMIT_SCALE * dy * dy;
  l.z = COMMIT_SCALE * dz * dz;
  l.w = COMMIT_SCALE * dw * dw;
  ((float4*)outL)[i] = l;
  if (off4 == 0) outIdx[n] = (float)idx;
}

// ---------------------------------------------------------------------------
// Host launch
// ---------------------------------------------------------------------------
extern "C" void kernel_launch(void* const* d_in, const int* in_sizes, int n_in,
                              void* d_out, int out_size, void* d_ws, size_t ws_size,
                              hipStream_t stream) {
  (void)in_sizes; (void)n_in; (void)out_size; (void)ws_size;

  const float* x = (const float*)d_in[0];   // [N, D] f32
  const float* c = (const float*)d_in[1];   // [K, D] f32
  // d_in[2] = train flag (0) — unused

  // Output layout (flat, return order): quantized_out, quantization_loss,
  // nn_idx, codebook_values, counts.
  float* outQ   = (float*)d_out;
  float* outL   = outQ + (size_t)VQ_N * VQ_D;
  float* outIdx = outL + (size_t)VQ_N * VQ_D;
  float* outCB  = outIdx + VQ_N;
  float* outCnt = outCB + (size_t)VQ_K * VQ_D;

  // Workspace layout (256B aligned slices).
  char* ws = (char*)d_ws;
  const size_t xbOff  = 0;
  const size_t cbOff  = (xbOff + (size_t)VQ_N * VQ_D * 2 + 255) & ~(size_t)255;
  const size_t cnOff  = (cbOff + (size_t)VQ_K * VQ_D * 2 + 255) & ~(size_t)255;
  const size_t idxOff = (cnOff + (size_t)VQ_K * 4 + 255) & ~(size_t)255;
  bf16*  xb      = (bf16*)(ws + xbOff);
  bf16*  cb      = (bf16*)(ws + cbOff);
  float* cnorm   = (float*)(ws + cnOff);
  int*   bestIdx = (int*)(ws + idxOff);

  // 1) Conversions (bandwidth-trivial vs the 275-GFLOP matmul).
  vq_cvt_x_kernel<<<(VQ_N * (VQ_D / 4)) / 256, 256, 0, stream>>>(x, xb);
  vq_cvt_c_kernel<<<(VQ_K * (VQ_D / 4)) / 256, 256, 0, stream>>>(c, cb, outCB);

  // 2) Centroid squared norms + counts output.
  vq_cnorm_kernel<<<VQ_K / 8, 256, 0, stream>>>(c, cnorm, outCnt);

  // 3) WMMA streaming argmin GEMM with async-LDS B staging (~303KB LDS).
  vq_argmin_kernel<<<VQ_N / XT_ROWS, 256, SMEM_BYTES, stream>>>(xb, cb, cnorm, bestIdx);

  // 4) Gather + elementwise outputs.
  vq_finalize_kernel<<<(VQ_N * (VQ_D / 4)) / 256, 256, 0, stream>>>(
      x, c, bestIdx, outQ, outL, outIdx);
}
